// MultiHeadAttention_81913616270105
// MI455X (gfx1250) — compile-verified, matrix operand order checked by reading, not run on hardware
//
#include <hip/hip_runtime.h>
#include <hip/hip_bf16.h>

// ---------------------------------------------------------------------------
// MI455X (gfx1250) pre-LN multi-head attention block, bf16 WMMA pipeline.
//   B=2, S=2048, D=1024, H=8, DH=128.  All matmuls on v_wmma_f32_16x16x32_bf16.
//   v6: fragment loads hoisted into dedicated registers ahead of each WMMA
//       block (kills accumulator v_mov/v_nop churn); async-LDS staging,
//       launch_bounds(256,2), conflict-free LDS pitches as in v5.
// ---------------------------------------------------------------------------

typedef __bf16 bf16_t;
typedef __attribute__((ext_vector_type(16))) __bf16 v16bf;
typedef __attribute__((ext_vector_type(8)))  float  v8f;
typedef __attribute__((ext_vector_type(4)))  float  v4f;

#define BB   2
#define SS   2048
#define DD   1024
#define HH   8
#define DH   128
#define BH   (BB * HH)      // 16
#define MM   (BB * SS)      // 4096 rows in flat [M, D] layout
#define KSTEPS (DD / 32)    // 32 K-steps per GEMM
#define NSTEPS (SS / 32)    // 64 K-steps in attention

static __device__ __forceinline__ v8f wmma_bf16(v16bf a, v16bf b, v8f c) {
    return __builtin_amdgcn_wmma_f32_16x16x32_bf16(
        /*neg_a=*/false, a, /*neg_b=*/false, b,
        /*c_mod=*/(short)0, c, /*reuse_a=*/false, /*reuse_b=*/false);
}

// Async global->LDS copy of 32 contiguous bytes per lane (2x B128).
// VDST = per-lane LDS byte address (low 32 bits of generic pointer = AS(3)
// offset); INST_OFFSET is added to both the LDS and global addresses.
static __device__ __forceinline__ void async_b128x2(void* lds, const void* g) {
    const unsigned l = (unsigned)(uintptr_t)lds;
    asm volatile("global_load_async_to_lds_b128 %0, %1, off\n\t"
                 "global_load_async_to_lds_b128 %0, %1, off offset:16"
                 :: "v"(l), "v"(g) : "memory");
}
static __device__ __forceinline__ void wait_async0() {
    asm volatile("s_wait_asynccnt 0x0" ::: "memory");
}

// ---------------------------------------------------------------------------
// 1) LayerNorm over D=1024, emit bf16 activations.
// ---------------------------------------------------------------------------
__global__ void ln_bf16_kernel(const float* __restrict__ x,
                               const float* __restrict__ gamma,
                               const float* __restrict__ beta,
                               bf16_t* __restrict__ Xb) {
    const int row  = blockIdx.x;
    const int tid  = threadIdx.x;
    const int wave = tid >> 5, lane = tid & 31;
    const float* xr = x + (size_t)row * DD;

    float v[4], s = 0.f, s2 = 0.f;
#pragma unroll
    for (int i = 0; i < 4; ++i) {
        float t = xr[i * 256 + tid];
        v[i] = t; s += t; s2 += t * t;
    }
#pragma unroll
    for (int off = 16; off >= 1; off >>= 1) {
        s  += __shfl_xor(s,  off, 32);
        s2 += __shfl_xor(s2, off, 32);
    }
    __shared__ float sh1[8], sh2[8];
    if (lane == 0) { sh1[wave] = s; sh2[wave] = s2; }
    __syncthreads();
    s = 0.f; s2 = 0.f;
#pragma unroll
    for (int w = 0; w < 8; ++w) { s += sh1[w]; s2 += sh2[w]; }
    const float mu   = s * (1.f / DD);
    const float var  = s2 * (1.f / DD) - mu * mu;
    const float rstd = rsqrtf(var + 1e-5f);
#pragma unroll
    for (int i = 0; i < 4; ++i) {
        int col = i * 256 + tid;
        Xb[(size_t)row * DD + col] =
            (bf16_t)((v[i] - mu) * rstd * gamma[col] + beta[col]);
    }
}

// ---------------------------------------------------------------------------
// 2) fp32 -> bf16 weight conversion (grid-strided).
// ---------------------------------------------------------------------------
__global__ void f2b_kernel(const float* __restrict__ W, bf16_t* __restrict__ Wb, int n) {
    for (int i = blockIdx.x * blockDim.x + threadIdx.x; i < n;
         i += gridDim.x * blockDim.x)
        Wb[i] = (bf16_t)W[i];
}

// ---------------------------------------------------------------------------
// 2b) One-shot V transpose: Vt[bh][d][s] = Vb[bh][s][d] (32x32 LDS tiles,
//     coalesced load and store).
// ---------------------------------------------------------------------------
__global__ void transpose_v_kernel(const bf16_t* __restrict__ Vb,
                                   bf16_t* __restrict__ Vt) {
    __shared__ bf16_t tile[32][33];
    const int bh = blockIdx.z;
    const int s0 = blockIdx.x * 32;
    const int d0 = blockIdx.y * 32;
    const int tid = threadIdx.x;           // 256
    const int r  = tid >> 3;               // 0..31
    const int c4 = (tid & 7) * 4;          // 0..28
    const bf16_t* src = Vb + ((size_t)bh * SS + s0 + r) * DH + d0 + c4;
#pragma unroll
    for (int j = 0; j < 4; ++j) tile[r][c4 + j] = src[j];
    __syncthreads();
    bf16_t* dst = Vt + ((size_t)bh * DH + d0 + r) * SS + s0 + c4;
#pragma unroll
    for (int j = 0; j < 4; ++j) dst[j] = tile[c4 + j][r];
}

// ---------------------------------------------------------------------------
// 3) Projection GEMM: dst = Xb @ W^T + bias  (bf16 in, fp32 accum, bf16 out).
//    Block = 256 threads = 8 waves; block tile 128x128; wave tile 64x32.
//    Double-buffered async-LDS staging; fragments loaded up-front into
//    dedicated registers, then 8 in-place WMMAs (no accumulator moves).
// ---------------------------------------------------------------------------
__global__ void __launch_bounds__(256, 2)
proj_gemm_kernel(const bf16_t* __restrict__ A,
                 const bf16_t* __restrict__ W,
                 const float*  __restrict__ bias,
                 bf16_t* __restrict__ dst) {
    __shared__ bf16_t Ast[2][128][40];   // 20 KB
    __shared__ bf16_t Wst[2][128][40];   // 20 KB

    const int tid  = threadIdx.x;
    const int lane = tid & 31, wave = tid >> 5;
    const int hi = lane >> 4, lo = lane & 15;
    const int mblk = blockIdx.y * 128, nblk = blockIdx.x * 128;
    const int ml = (wave & 1) * 64;      // wave M base within tile
    const int nl = (wave >> 1) * 32;     // wave N base within tile

    // cooperative staging map: 256 threads cover a 128x32 bf16 slab (32B each)
    const int crow = tid >> 1, ccol = (tid & 1) * 16;
    const bf16_t* pA = A + (size_t)(mblk + crow) * DD + ccol;
    const bf16_t* pW = W + (size_t)(nblk + crow) * DD + ccol;

    // prologue: stage slab 0 asynchronously
    async_b128x2(&Ast[0][crow][ccol], pA);
    async_b128x2(&Wst[0][crow][ccol], pW);
    pA += 32; pW += 32;

    v8f acc[4][2] = {};
    auto gemm_step = [&](int buf) {
        // hoist all six fragments into dedicated registers first
        const v16bf b0 = *(const v16bf*)&Wst[buf][nl + lo][hi * 16];
        const v16bf b1 = *(const v16bf*)&Wst[buf][nl + 16 + lo][hi * 16];
        const v16bf a0 = *(const v16bf*)&Ast[buf][ml +  0 + lo][hi * 16];
        const v16bf a1 = *(const v16bf*)&Ast[buf][ml + 16 + lo][hi * 16];
        const v16bf a2 = *(const v16bf*)&Ast[buf][ml + 32 + lo][hi * 16];
        const v16bf a3 = *(const v16bf*)&Ast[buf][ml + 48 + lo][hi * 16];
        acc[0][0] = wmma_bf16(a0, b0, acc[0][0]);
        acc[1][0] = wmma_bf16(a1, b0, acc[1][0]);
        acc[2][0] = wmma_bf16(a2, b0, acc[2][0]);
        acc[3][0] = wmma_bf16(a3, b0, acc[3][0]);
        acc[0][1] = wmma_bf16(a0, b1, acc[0][1]);
        acc[1][1] = wmma_bf16(a1, b1, acc[1][1]);
        acc[2][1] = wmma_bf16(a2, b1, acc[2][1]);
        acc[3][1] = wmma_bf16(a3, b1, acc[3][1]);
    };

    for (int s = 0; s < KSTEPS; ++s) {
        const int buf = s & 1;
        wait_async0();                       // slab[buf] copies (ours) complete
        __syncthreads();                     // ...and everyone else's
        if (s + 1 < KSTEPS) {                // overlap next slab with compute
            __builtin_prefetch(pA + 32, 0, 3);
            async_b128x2(&Ast[buf ^ 1][crow][ccol], pA);
            async_b128x2(&Wst[buf ^ 1][crow][ccol], pW);
            pA += 32; pW += 32;
        }
        gemm_step(buf);
    }

#pragma unroll
    for (int i = 0; i < 4; ++i)
#pragma unroll
        for (int t = 0; t < 2; ++t)
#pragma unroll
            for (int r = 0; r < 8; ++r) {
                const int row = mblk + ml + i * 16 + r + 8 * hi; // M = r + 8*(lane/16)
                const int col = nblk + nl + t * 16 + lo;         // N = lane%16
                dst[(size_t)row * DD + col] = (bf16_t)(acc[i][t][r] + bias[col]);
            }
}

// ---------------------------------------------------------------------------
// 4) Flash attention per (b,h): energy = Q K^T, online softmax, context = P V,
//    epilogue scale 1/sqrt(D).  Wave owns 16 q-rows; K tile (32x128, pitch
//    136 -> conflict-free) and Vt tile (128x32, pitch 40) staged by async-LDS
//    copies, double-buffered, one barrier per step.  Fragment arrays loaded
//    ahead of each WMMA block.
// ---------------------------------------------------------------------------
__global__ void __launch_bounds__(256, 2)
attn_kernel(const bf16_t* __restrict__ Qb,
            const bf16_t* __restrict__ Kb,
            const bf16_t* __restrict__ Vt,
            bf16_t* __restrict__ Cb) {
    __shared__ bf16_t Klds[2][32][136];    // 17 KB (pitch 136: 4r mod 64 banks)
    __shared__ bf16_t Vlds[2][DH][40];     // 20 KB (transposed V, pitch 40)
    __shared__ float  Pst[8][16][36];      // 18 KB, per-wave P strip

    const int tid  = threadIdx.x;
    const int lane = tid & 31, wave = tid >> 5;
    const int hi = lane >> 4, lo = lane & 15;
    const int bh = blockIdx.y;
    const int q0 = blockIdx.x * 128 + wave * 16;

    const bf16_t* Qh  = Qb + (size_t)bh * SS * DH;
    const bf16_t* Kh  = Kb + (size_t)bh * SS * DH;
    const bf16_t* Vth = Vt + (size_t)bh * DH * SS;

    // staging maps (32B per thread each)
    const int srow = tid >> 3, scol = (tid & 7) * 16;   // K: 32 rows x 128
    const int vrow = tid >> 1, vcol = (tid & 1) * 16;   // Vt: 128 rows x 32
    const bf16_t* kp  = Kh  + (size_t)srow * DH + scol;
    const bf16_t* vtp = Vth + (size_t)vrow * SS + vcol;

    // prologue: stage tile 0 asynchronously
    async_b128x2(&Klds[0][srow][scol], kp);
    async_b128x2(&Vlds[0][vrow][vcol], vtp);
    kp += 32 * DH; vtp += 32;

    v16bf qf[4];
#pragma unroll
    for (int c = 0; c < 4; ++c)
        qf[c] = *(const v16bf*)(Qh + (size_t)(q0 + lo) * DH + c * 32 + hi * 16);

    float m_i[8], l_i[8];
    v8f acc[8] = {};
#pragma unroll
    for (int r = 0; r < 8; ++r) { m_i[r] = -1e30f; l_i[r] = 0.f; }

    auto attn_step = [&](int buf) {
        // ---- energy tiles (16q x 32k), GEMM-K = DH = 128 ----
        v16bf kfr[8];
#pragma unroll
        for (int c = 0; c < 4; ++c) {
            kfr[2 * c]     = *(const v16bf*)&Klds[buf][lo][c * 32 + hi * 16];
            kfr[2 * c + 1] = *(const v16bf*)&Klds[buf][16 + lo][c * 32 + hi * 16];
        }
        v8f e0 = {}, e1 = {};
#pragma unroll
        for (int c = 0; c < 4; ++c) {
            e0 = wmma_bf16(qf[c], kfr[2 * c],     e0);
            e1 = wmma_bf16(qf[c], kfr[2 * c + 1], e1);
        }
        // ---- online softmax (row stats live across the 16 lanes of a half) ----
        float alpha[8];
#pragma unroll
        for (int r = 0; r < 8; ++r) {
            float mx = fmaxf(e0[r], e1[r]);
#pragma unroll
            for (int off = 8; off >= 1; off >>= 1)
                mx = fmaxf(mx, __shfl_xor(mx, off, 16));
            const float mn = fmaxf(m_i[r], mx);
            const float p0 = __expf(e0[r] - mn);
            const float p1 = __expf(e1[r] - mn);
            float srw = p0 + p1;
#pragma unroll
            for (int off = 8; off >= 1; off >>= 1)
                srw += __shfl_xor(srw, off, 16);
            alpha[r] = __expf(m_i[r] - mn);
            l_i[r]   = l_i[r] * alpha[r] + srw;
            m_i[r]   = mn;
            Pst[wave][r + 8 * hi][lo]      = p0;   // D-layout -> row-major strip
            Pst[wave][r + 8 * hi][16 + lo] = p1;
        }
#pragma unroll
        for (int n = 0; n < 8; ++n)
#pragma unroll
            for (int r = 0; r < 8; ++r) acc[n][r] *= alpha[r];

        // per-wave strip: LDS ops are in-order within a wave
        asm volatile("s_wait_dscnt 0x0" ::: "memory");

        // ---- rebuild P as an A-fragment (lane L: row L%16, k-chunk (L/16)*16) ----
        v16bf pf;
#pragma unroll
        for (int g = 0; g < 4; ++g) {
            const v4f pv = *(const v4f*)&Pst[wave][lo][hi * 16 + g * 4];
#pragma unroll
            for (int j = 0; j < 4; ++j)
                pf[g * 4 + j] = (bf16_t)pv[j];
        }
        // ---- context accumulation: acc += P @ V (transposed tile in LDS) ----
        v16bf vfr[8];
#pragma unroll
        for (int n = 0; n < 8; ++n)
            vfr[n] = *(const v16bf*)&Vlds[buf][n * 16 + lo][hi * 16];
#pragma unroll
        for (int n = 0; n < 8; ++n)
            acc[n] = wmma_bf16(pf, vfr[n], acc[n]);
    };

    for (int s = 0; s < NSTEPS; ++s) {
        const int buf = s & 1;
        wait_async0();                       // tile[buf] copies (ours) complete
        __syncthreads();                     // ...and everyone else's
        if (s + 1 < NSTEPS) {                // overlap next tile with compute
            __builtin_prefetch(kp + 32 * DH, 0, 3);
            async_b128x2(&Klds[buf ^ 1][srow][scol], kp);
            async_b128x2(&Vlds[buf ^ 1][vrow][vcol], vtp);
            kp += 32 * DH; vtp += 32;
        }
        attn_step(buf);
    }

    // ---- epilogue: 1/l * 1/sqrt(D) ----
    const float sc = 0.03125f;   // 1/sqrt(1024)
#pragma unroll
    for (int n = 0; n < 8; ++n)
#pragma unroll
        for (int r = 0; r < 8; ++r) {
            const int q = q0 + r + 8 * hi;
            const int d = n * 16 + lo;
            Cb[((size_t)bh * SS + q) * DH + d] = (bf16_t)(acc[n][r] / l_i[r] * sc);
        }
}

// ---------------------------------------------------------------------------
// 5) Output projection + bias + residual, fp32 out.  Same async-LDS-staged
//    GEMM core as the projections.
// ---------------------------------------------------------------------------
__global__ void __launch_bounds__(256, 2)
out_proj_kernel(const bf16_t* __restrict__ A,
                const bf16_t* __restrict__ W,
                const float*  __restrict__ bias,
                const float*  __restrict__ resid,
                float* __restrict__ out) {
    __shared__ bf16_t Ast[2][128][40];
    __shared__ bf16_t Wst[2][128][40];

    const int tid  = threadIdx.x;
    const int lane = tid & 31, wave = tid >> 5;
    const int hi = lane >> 4, lo = lane & 15;
    const int mblk = blockIdx.y * 128, nblk = blockIdx.x * 128;
    const int ml = (wave & 1) * 64;
    const int nl = (wave >> 1) * 32;

    const int crow = tid >> 1, ccol = (tid & 1) * 16;
    const bf16_t* pA = A + (size_t)(mblk + crow) * DD + ccol;
    const bf16_t* pW = W + (size_t)(nblk + crow) * DD + ccol;

    async_b128x2(&Ast[0][crow][ccol], pA);
    async_b128x2(&Wst[0][crow][ccol], pW);
    pA += 32; pW += 32;

    v8f acc[4][2] = {};
    auto gemm_step = [&](int buf) {
        const v16bf b0 = *(const v16bf*)&Wst[buf][nl + lo][hi * 16];
        const v16bf b1 = *(const v16bf*)&Wst[buf][nl + 16 + lo][hi * 16];
        const v16bf a0 = *(const v16bf*)&Ast[buf][ml +  0 + lo][hi * 16];
        const v16bf a1 = *(const v16bf*)&Ast[buf][ml + 16 + lo][hi * 16];
        const v16bf a2 = *(const v16bf*)&Ast[buf][ml + 32 + lo][hi * 16];
        const v16bf a3 = *(const v16bf*)&Ast[buf][ml + 48 + lo][hi * 16];
        acc[0][0] = wmma_bf16(a0, b0, acc[0][0]);
        acc[1][0] = wmma_bf16(a1, b0, acc[1][0]);
        acc[2][0] = wmma_bf16(a2, b0, acc[2][0]);
        acc[3][0] = wmma_bf16(a3, b0, acc[3][0]);
        acc[0][1] = wmma_bf16(a0, b1, acc[0][1]);
        acc[1][1] = wmma_bf16(a1, b1, acc[1][1]);
        acc[2][1] = wmma_bf16(a2, b1, acc[2][1]);
        acc[3][1] = wmma_bf16(a3, b1, acc[3][1]);
    };

    for (int s = 0; s < KSTEPS; ++s) {
        const int buf = s & 1;
        wait_async0();
        __syncthreads();
        if (s + 1 < KSTEPS) {
            __builtin_prefetch(pA + 32, 0, 3);
            async_b128x2(&Ast[buf ^ 1][crow][ccol], pA);
            async_b128x2(&Wst[buf ^ 1][crow][ccol], pW);
            pA += 32; pW += 32;
        }
        gemm_step(buf);
    }

#pragma unroll
    for (int i = 0; i < 4; ++i)
#pragma unroll
        for (int t = 0; t < 2; ++t)
#pragma unroll
            for (int r = 0; r < 8; ++r) {
                const int row = mblk + ml + i * 16 + r + 8 * hi;
                const int col = nblk + nl + t * 16 + lo;
                out[(size_t)row * DD + col] =
                    acc[i][t][r] + bias[col] + resid[(size_t)row * DD + col];
            }
}

// ---------------------------------------------------------------------------
// Launch: LN -> weight cvt -> Q/K/V projections -> V transpose -> attention
//         -> O projection.  Workspace ~56 MB, all hot data L2-resident.
// ---------------------------------------------------------------------------
extern "C" void kernel_launch(void* const* d_in, const int* in_sizes, int n_in,
                              void* d_out, int out_size, void* d_ws, size_t ws_size,
                              hipStream_t stream) {
    const float* q     = (const float*)d_in[0];
    const float* Wq    = (const float*)d_in[1];
    const float* bq    = (const float*)d_in[2];
    const float* Wk    = (const float*)d_in[3];
    const float* bk    = (const float*)d_in[4];
    const float* Wv    = (const float*)d_in[5];
    const float* bv    = (const float*)d_in[6];
    const float* Wo    = (const float*)d_in[7];
    const float* bo    = (const float*)d_in[8];
    const float* gamma = (const float*)d_in[9];
    const float* beta  = (const float*)d_in[10];
    float* out = (float*)d_out;

    size_t off = 0;
    auto carve = [&](size_t bytes) -> void* {
        void* r = (char*)d_ws + off;
        off += (bytes + 255) & ~(size_t)255;
        return r;
    };
    bf16_t* Xb  = (bf16_t*)carve((size_t)MM * DD * sizeof(bf16_t));
    bf16_t* Wqb = (bf16_t*)carve((size_t)DD * DD * sizeof(bf16_t));
    bf16_t* Wkb = (bf16_t*)carve((size_t)DD * DD * sizeof(bf16_t));
    bf16_t* Wvb = (bf16_t*)carve((size_t)DD * DD * sizeof(bf16_t));
    bf16_t* Wob = (bf16_t*)carve((size_t)DD * DD * sizeof(bf16_t));
    bf16_t* Qb  = (bf16_t*)carve((size_t)MM * DD * sizeof(bf16_t));
    bf16_t* Kb  = (bf16_t*)carve((size_t)MM * DD * sizeof(bf16_t));
    bf16_t* Vb  = (bf16_t*)carve((size_t)MM * DD * sizeof(bf16_t));
    bf16_t* Vt  = (bf16_t*)carve((size_t)MM * DD * sizeof(bf16_t));
    bf16_t* Cb  = (bf16_t*)carve((size_t)MM * DD * sizeof(bf16_t));
    (void)ws_size; (void)in_sizes; (void)n_in; (void)out_size;

    // 1) LayerNorm -> bf16
    ln_bf16_kernel<<<MM, 256, 0, stream>>>(q, gamma, beta, Xb);

    // 2) weights -> bf16
    f2b_kernel<<<256, 256, 0, stream>>>(Wq, Wqb, DD * DD);
    f2b_kernel<<<256, 256, 0, stream>>>(Wk, Wkb, DD * DD);
    f2b_kernel<<<256, 256, 0, stream>>>(Wv, Wvb, DD * DD);
    f2b_kernel<<<256, 256, 0, stream>>>(Wo, Wob, DD * DD);

    // 3) Q/K/V projections (plain row-major bf16)
    dim3 gproj(DD / 128, MM / 128);
    proj_gemm_kernel<<<gproj, 256, 0, stream>>>(Xb, Wqb, bq, Qb);
    proj_gemm_kernel<<<gproj, 256, 0, stream>>>(Xb, Wkb, bk, Kb);
    proj_gemm_kernel<<<gproj, 256, 0, stream>>>(Xb, Wvb, bv, Vb);

    // 3b) one-shot V transpose for column-contiguous P@V fragments
    dim3 gtr(SS / 32, DH / 32, BH);
    transpose_v_kernel<<<gtr, 256, 0, stream>>>(Vb, Vt);

    // 4) flash attention per (b,h)
    dim3 gattn(SS / 128, BH);
    attn_kernel<<<gattn, 256, 0, stream>>>(Qb, Kb, Vt, Cb);

    // 5) output projection + residual
    out_proj_kernel<<<gproj, 256, 0, stream>>>(Cb, Wob, bo, q, out);
}
